// Micro_MIL_85512798863752
// MI455X (gfx1250) — matrix operand order repaired and checked by reading, not compiled, not run on hardware
//
#include <hip/hip_runtime.h>
#include <hip/hip_bf16.h>

// ---------------------------------------------------------------------------
// Micro_MIL forward on MI455X (gfx1250, wave32, WMMA + TDM).
// Dominant GEMM (16384x1024x1024) done in bf16 WMMA w/ f32 accumulate:
//   f32-WMMA (16x16x4) would be compute-bound (~8x fewer MACs/instr);
//   bf16 (16x16x32) puts the stage on the 23.3 TB/s HBM roofline (~6us),
//   with the 8x N-tile re-read of x absorbed by the 192 MB L2.
// Straight-through Gumbel hard == pure argmax in forward -> rep is a gather.
// B tiles are staged into LDS by the Tensor Data Mover (TENSORcnt), A tiles
// by the threads (they need the fp32->bf16 convert anyway).
// ---------------------------------------------------------------------------

typedef __bf16 bf16_t;
typedef __attribute__((ext_vector_type(16))) __bf16 v16bf;
typedef __attribute__((ext_vector_type(8)))  float  v8f;

#define N_INST 16384
#define DIM    1024
#define CNUM   36
#define CPAD   48
#define HDIM   128
#define KCLS   2

#define BM 128
#define BN 128
#define BK 32
#define LDA 40   // LDS row stride in bf16 elements: 32 data + 8 pad (80B, 16B-aligned)

__device__ __forceinline__ bf16_t f2bf(float f) {
  return (bf16_t)f;   // fptrunc f32->bf16, RNE; native convert on gfx1250
}

__device__ __forceinline__ float lrelu001(float v) { return v > 0.f ? v : 0.01f * v; }
__device__ __forceinline__ float lrelu02 (float v) { return v > 0.f ? v : 0.2f  * v; }

// Build a v16bf WMMA fragment from two 8-element (16B) chunks.
__device__ __forceinline__ v16bf make_frag(const bf16_t* p0, const bf16_t* p1) {
  v16bf r;
#pragma unroll
  for (int i = 0; i < 8; ++i) { r[i] = p0[i]; r[8 + i] = p1[i]; }
  return r;
}

__device__ __forceinline__ v8f wmma_bf16(v16bf a, v16bf b, v8f c) {
  // D = A(16x32 bf16) x B(32x16 bf16) + C(16x16 f32)
  return __builtin_amdgcn_wmma_f32_16x16x32_bf16(false, a, false, b, (short)0, c,
                                                 false, false);
}

// ---------------------------------------------------------------------------
// Tensor Data Mover: 2D bf16 tile -> LDS, with LDS row padding matching LDA.
// Descriptor per CDNA5 ISA ch.8 (group0: count/lds/global/type, group1: dims).
// ---------------------------------------------------------------------------
#if __has_builtin(__builtin_amdgcn_tensor_load_to_lds)
#define USE_TDM 1
typedef unsigned tdm_v4u __attribute__((ext_vector_type(4)));
typedef int      tdm_v8i __attribute__((ext_vector_type(8)));
typedef int      tdm_v4i __attribute__((ext_vector_type(4)));

__device__ __forceinline__ void tdm_load_tile_bf16(
    const bf16_t* gsrc,          // tile start in global memory
    unsigned lds_byte_addr,      // LDS destination byte address
    int tensor_d0, int tensor_d1,// remaining tensor extent (OOB clamp), elems
    int tile_k, int tile_rows,   // tile dims (dim0 contiguous), elems
    int row_stride)              // tensor_dim0_stride, elems
{
  const unsigned long long ga = (unsigned long long)(size_t)gsrc;
  tdm_v4u g0;
  g0[0] = 1u;                                        // count=1, user D#
  g0[1] = lds_byte_addr;
  g0[2] = (unsigned)(ga & 0xFFFFFFFFull);            // global_addr[31:0]
  g0[3] = (unsigned)((ga >> 32) & 0x1FFFFFFull)      // global_addr[56:32]
        | (2u << 30);                                // type=2 (image)
  tdm_v8i g1;
  g1[0] = (1 << 16)      // data_size = 2 bytes
        | (1 << 20)      // pad_enable: pad LDS rows
        | (3 << 22)      // pad_interval: 16 DWORDs (=64B data per row)
        | (3 << 25);     // pad_amount: 4 DWORDs (=16B -> 80B LDS row = LDA)
  g1[1] = (tensor_d0 & 0xFFFF) << 16;                              // dim0[15:0]
  g1[2] = ((tensor_d0 >> 16) & 0xFFFF) | ((tensor_d1 & 0xFFFF) << 16);
  g1[3] = ((tensor_d1 >> 16) & 0xFFFF) | ((tile_k & 0xFFFF) << 16); // tile_dim0
  g1[4] = (tile_rows & 0xFFFF);                      // tile_dim1 (tile_dim2=0)
  g1[5] = row_stride;                                // dim0_stride[31:0]
  g1[6] = 0;
  g1[7] = 0;
  tdm_v4i gz = {0, 0, 0, 0};
#if __clang_major__ >= 23
  tdm_v8i gz8 = {0, 0, 0, 0, 0, 0, 0, 0};
  __builtin_amdgcn_tensor_load_to_lds(g0, g1, gz, gz, gz8, 0);
#else
  __builtin_amdgcn_tensor_load_to_lds(g0, g1, gz, gz, 0);
#endif
}
#else
#define USE_TDM 0
#endif

// ---------------------------------------------------------------------------
// K0: prep. blocks [0,1024): W_img^T -> bf16 (1024x1024, 32x32 LDS tiles)
//           blocks [1024,1152): W1^T fp32 (128x1024)
//           blocks [1152,1200): centers -> bf16 rows (zero-padded to 48) + ||c||^2
// ---------------------------------------------------------------------------
__global__ __launch_bounds__(256)
void k0_prep(const float* __restrict__ W_img,
             const float* __restrict__ W1,
             const float* __restrict__ centers,
             bf16_t* __restrict__ Wt,      // [DIM][DIM], Wt[n][k] = W_img[k][n]
             float*  __restrict__ W1t,     // [HDIM][DIM]
             bf16_t* __restrict__ cbf,     // [CPAD][DIM]
             float*  __restrict__ cnorm)   // [CPAD]
{
  const int t = threadIdx.x;
  int b = blockIdx.x;
  if (b < 1024) {
    __shared__ float tile[32][33];
    const int ti = b >> 5, tj = b & 31;
    const int tx = t & 31, ty0 = t >> 5;
#pragma unroll
    for (int k = 0; k < 4; ++k) {
      const int ty = ty0 + k * 8;
      tile[ty][tx] = W_img[(size_t)(ti * 32 + ty) * DIM + tj * 32 + tx];
    }
    __syncthreads();
#pragma unroll
    for (int k = 0; k < 4; ++k) {
      const int ty = ty0 + k * 8;
      Wt[(size_t)(tj * 32 + ty) * DIM + ti * 32 + tx] = f2bf(tile[tx][ty]);
    }
  } else if (b < 1024 + 128) {
    b -= 1024;
    __shared__ float tile[32][33];
    const int ti = b >> 2, tj = b & 3;   // ti over k (32 tiles), tj over h (4 tiles)
    const int tx = t & 31, ty0 = t >> 5;
#pragma unroll
    for (int k = 0; k < 4; ++k) {
      const int ty = ty0 + k * 8;
      tile[ty][tx] = W1[(size_t)(ti * 32 + ty) * HDIM + tj * 32 + tx];
    }
    __syncthreads();
#pragma unroll
    for (int k = 0; k < 4; ++k) {
      const int ty = ty0 + k * 8;
      W1t[(size_t)(tj * 32 + ty) * DIM + ti * 32 + tx] = tile[tx][ty];
    }
  } else {
    const int n = b - 1152;              // 0..47
    __shared__ float red[256];
    float s = 0.f;
    for (int k = t; k < DIM; k += 256) {
      const float v = (n < CNUM) ? centers[(size_t)n * DIM + k] : 0.f;
      cbf[(size_t)n * DIM + k] = f2bf(v);
      s += v * v;
    }
    red[t] = s;
    __syncthreads();
    for (int o = 128; o > 0; o >>= 1) {
      if (t < o) red[t] += red[t + o];
      __syncthreads();
    }
    if (t == 0) cnorm[n] = red[0];
  }
}

// ---------------------------------------------------------------------------
// K1: h = lrelu(x @ W_img + b_img), bf16 WMMA, BMxBNxBK = 128x128x32.
// 256 threads = 8 waves arranged 2(M) x 4(N); each wave: 64x32 = 4x2 WMMA accums.
// B tile staged by TDM (wave 0, TENSORcnt); A tile by threads (needs convert).
// ---------------------------------------------------------------------------
__global__ __launch_bounds__(256)
void k1_gemm_h(const float* __restrict__ x,
               const bf16_t* __restrict__ Wt,     // [DIM][DIM]
               const float* __restrict__ b_img,
               float* __restrict__ h)
{
  __shared__ bf16_t lA[BM * LDA];   // x tile, row-major [m][k]
  __shared__ bf16_t lB[BN * LDA];   // W^T tile, row-major [n][k]
  const int t = threadIdx.x;
  const int lane = t & 31, wid = t >> 5;
  const int waveM = wid & 1, waveN = wid >> 1;
  const int mBase = (blockIdx.x >> 3) * BM;   // 128 M tiles
  const int nBase = (blockIdx.x & 7) * BN;    // 8 N tiles

  v8f acc[4][2];
#pragma unroll
  for (int i = 0; i < 4; ++i)
#pragma unroll
    for (int j = 0; j < 2; ++j)
#pragma unroll
      for (int v = 0; v < 8; ++v) acc[i][j][v] = 0.f;

  const int arow = t >> 3;          // 0..31 (+32 per pass)
  const int acg  = (t & 7) * 4;     // 4-float column group
  const int hs = lane >> 4, lr = lane & 15;
#if USE_TDM
  const unsigned ldsB = (unsigned)(size_t)(void*)&lB[0];
#else
  const int brow  = t >> 1;         // 0..127
  const int bhalf = (t & 1) * 16;   // 16-bf16 half-row
#endif

  for (int kt = 0; kt < DIM; kt += BK) {
    // ---- stage B: W^T rows [nBase..nBase+127] x [kt..kt+31]
#if USE_TDM
    if (wid == 0) {
      tdm_load_tile_bf16(&Wt[(size_t)nBase * DIM + kt], ldsB,
                         DIM - kt, BN, BK, BN, DIM);
      __builtin_amdgcn_s_wait_tensorcnt((short)0);
    }
#else
    *(uint4*)&lB[brow * LDA + bhalf] =
        *(const uint4*)&Wt[(size_t)(nBase + brow) * DIM + kt + bhalf];
#endif
    // ---- stage A: fp32 -> bf16 (coalesced float4 reads)
#pragma unroll
    for (int p = 0; p < 4; ++p) {
      const int r = arow + p * 32;
      const float4 v4 = *(const float4*)&x[(size_t)(mBase + r) * DIM + kt + acg];
      bf16_t* dst = &lA[r * LDA + acg];
      dst[0] = f2bf(v4.x); dst[1] = f2bf(v4.y);
      dst[2] = f2bf(v4.z); dst[3] = f2bf(v4.w);
    }
    if (kt + BK < DIM)   // global_prefetch_b8 of next A tile
      __builtin_prefetch(&x[(size_t)(mBase + arow) * DIM + kt + BK + acg], 0, 1);
    __syncthreads();

    // ---- fragments per ISA 7.12.2 lane layouts
    v16bf af[4], bfrag[2];
#pragma unroll
    for (int i = 0; i < 4; ++i) {
      const bf16_t* p = &lA[(waveM * 64 + i * 16 + lr) * LDA + hs * 8];
      af[i] = make_frag(p, p + 16);           // K = {hs*8..+7} U {16+hs*8..+7}
    }
#pragma unroll
    for (int j = 0; j < 2; ++j) {
      const bf16_t* p = &lB[(waveN * 32 + j * 16 + lr) * LDA + hs * 8];
      bfrag[j] = make_frag(p, p + 16);        // K = {hs*8..+7} U {16+hs*8..+7}, N = lr
    }
#pragma unroll
    for (int i = 0; i < 4; ++i)
#pragma unroll
      for (int j = 0; j < 2; ++j)
        acc[i][j] = wmma_bf16(af[i], bfrag[j], acc[i][j]);
    __syncthreads();
  }

  // ---- epilogue: bias + leaky relu (0.01), fp32 store
#pragma unroll
  for (int j = 0; j < 2; ++j) {
    const int col = nBase + waveN * 32 + j * 16 + lr;
    const float bias = b_img[col];
#pragma unroll
    for (int i = 0; i < 4; ++i) {
      const int row0 = mBase + waveM * 64 + i * 16 + hs * 8;
#pragma unroll
      for (int v = 0; v < 8; ++v)
        h[(size_t)(row0 + v) * DIM + col] = lrelu001(acc[i][j][v] + bias);
    }
  }
}

// ---------------------------------------------------------------------------
// K2: per 128-row block: hc = h @ centers^T (WMMA, 3 padded N-tiles), fused
// ||h||^2 and h.w_attn accumulation, Student-t assign, scores (+g1), and the
// per-column block-local argmax partials.
// ---------------------------------------------------------------------------
__global__ __launch_bounds__(256)
void k2_scores(const float* __restrict__ h,
               const bf16_t* __restrict__ cbf,   // [CPAD][DIM]
               const float* __restrict__ cnorm,  // [CPAD]
               const float* __restrict__ w_attn,
               const float* __restrict__ b_attn, // scalar
               const float* __restrict__ g1,     // [N][C]
               float* __restrict__ pval,         // [CNUM][128]
               int*   __restrict__ pidx)         // [CNUM][128]
{
  __shared__ float sNum[128 * 40];
  __shared__ float sNrm[128];
  __shared__ float sAtt[128];
  const int t = threadIdx.x, lane = t & 31, wid = t >> 5;
  const int hs = lane >> 4, lr = lane & 15;
  const int rBase = blockIdx.x * 128;
  const int row = rBase + wid * 16 + lr;        // this lane's A-matrix row

  v8f acc[3];
#pragma unroll
  for (int j = 0; j < 3; ++j)
#pragma unroll
    for (int v = 0; v < 8; ++v) acc[j][v] = 0.f;

  float nrm = 0.f, att = 0.f;
  for (int k0 = 0; k0 < DIM; k0 += 32) {
    const float* hp = &h[(size_t)row * DIM + k0];
    v16bf a;
#pragma unroll
    for (int i = 0; i < 8; ++i) {
      const float v0 = hp[hs * 8 + i];
      const float v1 = hp[16 + hs * 8 + i];
      nrm += v0 * v0 + v1 * v1;
      att += v0 * w_attn[k0 + hs * 8 + i] + v1 * w_attn[k0 + 16 + hs * 8 + i];
      a[i] = f2bf(v0);
      a[8 + i] = f2bf(v1);
    }
#pragma unroll
    for (int j = 0; j < 3; ++j) {
      const bf16_t* p = &cbf[(size_t)(j * 16 + lr) * DIM + k0 + hs * 8];
      acc[j] = wmma_bf16(a, make_frag(p, p + 16), acc[j]);
    }
  }
  // lane pair (l, l^16) covers complementary K halves of the same row
  nrm += __shfl_xor(nrm, 16);
  att += __shfl_xor(att, 16);
  if (lane < 16) { sNrm[wid * 16 + lr] = nrm; sAtt[wid * 16 + lr] = att; }
  __syncthreads();

  // scatter num = 1/(1+d2) from accumulator layout (row = hs*8+v, col = lr+16j)
#pragma unroll
  for (int j = 0; j < 3; ++j) {
    const int col = j * 16 + lr;
    if (col < CNUM) {
      const float cn = cnorm[col];
#pragma unroll
      for (int v = 0; v < 8; ++v) {
        const int rl = wid * 16 + hs * 8 + v;
        const float d2 = sNrm[rl] - 2.f * acc[j][v] + cn;
        sNum[rl * 40 + col] = 1.f / (1.f + d2);
      }
    }
  }
  __syncthreads();

  // per-row normalize, score = assign * (h.w_attn) + b_attn + g1
  const float ba = b_attn[0];
  if (t < 128) {
    float s = 0.f;
    for (int c = 0; c < CNUM; ++c) s += sNum[t * 40 + c];
    const float inv = 1.f / s;
    const float at = sAtt[t];
    for (int c = 0; c < CNUM; ++c)
      sNum[t * 40 + c] = sNum[t * 40 + c] * inv * at + ba +
                         g1[(size_t)(rBase + t) * CNUM + c];
  }
  __syncthreads();

  // per-column block argmax
  if (t < CNUM) {
    float best = -3.4e38f;
    int bi = rBase;
    for (int r = 0; r < 128; ++r) {
      const float v = sNum[r * 40 + t];
      if (v > best) { best = v; bi = rBase + r; }
    }
    pval[t * gridDim.x + blockIdx.x] = best;
    pidx[t * gridDim.x + blockIdx.x] = bi;
  }
}

// ---------------------------------------------------------------------------
// K3: final per-column argmax over the 128 block partials.
// ---------------------------------------------------------------------------
__global__ void k3_argmax(const float* __restrict__ pval,
                          const int* __restrict__ pidx,
                          int nblk, int* __restrict__ idxOut)
{
  const int t = threadIdx.x;
  if (t < CNUM) {
    float best = -3.4e38f;
    int bi = 0;
    for (int b = 0; b < nblk; ++b) {
      const float v = pval[t * nblk + b];
      if (v > best) { best = v; bi = pidx[t * nblk + b]; }
    }
    idxOut[t] = bi;
  }
}

// ---------------------------------------------------------------------------
// K4: rep[c] = h[idx[c]]; row norms; f1 = rep @ W1 (via W1^T rows). 36 blocks.
// ---------------------------------------------------------------------------
__global__ __launch_bounds__(256)
void k4_rep_f1(const float* __restrict__ h,
               const int* __restrict__ idx,
               const float* __restrict__ W1t,  // [HDIM][DIM]
               float* __restrict__ rep,        // [CNUM][DIM]
               float* __restrict__ repn,       // [CNUM]
               float* __restrict__ f1)         // [CNUM][HDIM]
{
  __shared__ float sRep[DIM];
  __shared__ float red[256];
  const int c = blockIdx.x, t = threadIdx.x;
  const size_t rrow = (size_t)idx[c] * DIM;
  float ns = 0.f;
  for (int k = t; k < DIM; k += 256) {
    const float v = h[rrow + k];
    sRep[k] = v;
    rep[(size_t)c * DIM + k] = v;
    ns += v * v;
  }
  red[t] = ns;
  __syncthreads();
  for (int o = 128; o > 0; o >>= 1) {
    if (t < o) red[t] += red[t + o];
    __syncthreads();
  }
  if (t == 0) repn[c] = sqrtf(red[0]);

  // f1[c][j]: two threads per output, 512-element halves, combined via shfl
  const int j = t >> 1, half = t & 1;
  const float* wp = &W1t[(size_t)j * DIM + half * 512];
  const float* rp = &sRep[half * 512];
  float s = 0.f;
  for (int k = 0; k < 512; ++k) s += rp[k] * wp[k];
  s += __shfl_xor(s, 1);
  if (half == 0) f1[(size_t)c * HDIM + j] = s;
}

// ---------------------------------------------------------------------------
// K5: cosine sim + edge argmax, two exact GAT layers (count-weighted softmax,
// duplicate self-loop multiplicity), mean pool, classifier. Single block.
// ---------------------------------------------------------------------------
__device__ void gat_layer(const float* __restrict__ f, float* __restrict__ outbuf,
                          float* __restrict__ alpha, float* __restrict__ el,
                          float* __restrict__ er, const int* __restrict__ amax,
                          const float* __restrict__ al, const float* __restrict__ ar,
                          const float* __restrict__ bvec)
{
  const int t = threadIdx.x;
  if (t < CNUM) {
    float sl = 0.f, sr = 0.f;
    for (int k = 0; k < HDIM; ++k) {
      const float fv = f[t * HDIM + k];
      sl += fv * al[k];
      sr += fv * ar[k];
    }
    el[t] = sl;
    er[t] = sr;
  }
  __syncthreads();
  if (t < CNUM) {                 // softmax over in-edges i for dst column j = t
    const int j = t;
    float m = -3.4e38f;
    for (int i = 0; i < CNUM; ++i) {
      const int mult = (amax[i] == j) + (i == j);
      if (mult > 0) m = fmaxf(m, lrelu02(el[i] + er[j]));
    }
    float s = 0.f;
    for (int i = 0; i < CNUM; ++i) {
      const int mult = (amax[i] == j) + (i == j);
      const float e =
          (mult > 0) ? (float)mult * __expf(lrelu02(el[i] + er[j]) - m) : 0.f;
      alpha[i * CNUM + j] = e;
      s += e;
    }
    const float inv = 1.f / s;
    for (int i = 0; i < CNUM; ++i) alpha[i * CNUM + j] *= inv;
  }
  __syncthreads();
  for (int p = t; p < CNUM * HDIM; p += 256) {
    const int j = p / HDIM, d = p % HDIM;
    float s = 0.f;
    for (int i = 0; i < CNUM; ++i) s += alpha[i * CNUM + j] * f[i * HDIM + d];
    outbuf[p] = lrelu001(s + bvec[d]);   // lrelu fused with gat output
  }
  __syncthreads();
}

__global__ __launch_bounds__(256)
void k5_graph(const float* __restrict__ rep, const float* __restrict__ repn,
              const float* __restrict__ g2, const float* __restrict__ f1g,
              const float* __restrict__ al1, const float* __restrict__ ar1,
              const float* __restrict__ b1, const float* __restrict__ W2,
              const float* __restrict__ al2, const float* __restrict__ ar2,
              const float* __restrict__ b2, const float* __restrict__ Wc,
              const float* __restrict__ bc, float* __restrict__ out)
{
  __shared__ float f[CNUM * HDIM];
  __shared__ float hbuf[CNUM * HDIM];
  __shared__ float alpha[CNUM * CNUM];
  __shared__ float el[CNUM], er[CNUM];
  __shared__ int amax[CNUM];
  __shared__ float wsi[HDIM];
  const int t = threadIdx.x;

  // cosine similarity + g2 (alpha reused as sim scratch), all 256 threads
  for (int p = t; p < CNUM * CNUM; p += 256) {
    const int i = p / CNUM, j = p % CNUM;
    float d = 0.f;
    for (int k = 0; k < DIM; ++k)
      d += rep[(size_t)i * DIM + k] * rep[(size_t)j * DIM + k];
    alpha[p] = d / (fmaxf(repn[i], 1e-8f) * fmaxf(repn[j], 1e-8f)) + g2[p];
  }
  for (int p = t; p < CNUM * HDIM; p += 256) f[p] = f1g[p];
  __syncthreads();
  if (t < CNUM) {   // row-wise edge argmax
    float best = -3.4e38f;
    int bj = 0;
    for (int j = 0; j < CNUM; ++j) {
      const float v = alpha[t * CNUM + j];
      if (v > best) { best = v; bj = j; }
    }
    amax[t] = bj;
  }
  __syncthreads();

  gat_layer(f, hbuf, alpha, el, er, amax, al1, ar1, b1);   // h1 (lrelu fused)

  // f2 = h1 @ W2
  for (int p = t; p < CNUM * HDIM; p += 256) {
    const int c = p / HDIM, j = p % HDIM;
    float s = 0.f;
    for (int k = 0; k < HDIM; ++k) s += hbuf[c * HDIM + k] * W2[k * HDIM + j];
    f[p] = s;
  }
  __syncthreads();

  gat_layer(f, hbuf, alpha, el, er, amax, al2, ar2, b2);   // h2 (lrelu fused)

  if (t < HDIM) {
    float s = 0.f;
    for (int c = 0; c < CNUM; ++c) s += hbuf[c * HDIM + t];
    wsi[t] = s / (float)CNUM;
  }
  __syncthreads();
  if (t < KCLS) {
    float s = bc[t];
    for (int d = 0; d < HDIM; ++d) s += wsi[d] * Wc[d * KCLS + t];
    out[t] = lrelu001(s);
  }
}

// ---------------------------------------------------------------------------
extern "C" void kernel_launch(void* const* d_in, const int* in_sizes, int n_in,
                              void* d_out, int out_size, void* d_ws, size_t ws_size,
                              hipStream_t stream) {
  (void)in_sizes; (void)n_in; (void)out_size; (void)ws_size;
  const float* x       = (const float*)d_in[0];
  const float* g1      = (const float*)d_in[1];
  const float* g2      = (const float*)d_in[2];
  const float* W_img   = (const float*)d_in[3];
  const float* b_img   = (const float*)d_in[4];
  const float* centers = (const float*)d_in[5];
  const float* w_attn  = (const float*)d_in[6];
  const float* b_attn  = (const float*)d_in[7];
  const float* W1      = (const float*)d_in[8];
  const float* al1     = (const float*)d_in[9];
  const float* ar1     = (const float*)d_in[10];
  const float* b1      = (const float*)d_in[11];
  const float* W2      = (const float*)d_in[12];
  const float* al2     = (const float*)d_in[13];
  const float* ar2     = (const float*)d_in[14];
  const float* b2      = (const float*)d_in[15];
  const float* Wc      = (const float*)d_in[16];
  const float* bc      = (const float*)d_in[17];

  char* ws = (char*)d_ws;
  size_t off = 0;
  auto alloc = [&](size_t bytes) -> void* {
    void* p = ws + off;
    off = (off + bytes + 255) & ~(size_t)255;
    return p;
  };
  float*  h     = (float*) alloc((size_t)N_INST * DIM * 4);   // 64 MB
  bf16_t* Wt    = (bf16_t*)alloc((size_t)DIM * DIM * 2);      //  2 MB
  bf16_t* cbf   = (bf16_t*)alloc((size_t)CPAD * DIM * 2);
  float*  cnorm = (float*) alloc(CPAD * 4);
  float*  W1t   = (float*) alloc((size_t)HDIM * DIM * 4);
  float*  pval  = (float*) alloc((size_t)CNUM * 128 * 4);
  int*    pidx  = (int*)   alloc((size_t)CNUM * 128 * 4);
  int*    idx   = (int*)   alloc(CNUM * 4);
  float*  rep   = (float*) alloc((size_t)CNUM * DIM * 4);
  float*  repn  = (float*) alloc(CNUM * 4);
  float*  f1    = (float*) alloc((size_t)CNUM * HDIM * 4);

  k0_prep   <<<1200, 256, 0, stream>>>(W_img, W1, centers, Wt, W1t, cbf, cnorm);
  k1_gemm_h <<<(N_INST / BM) * (DIM / BN), 256, 0, stream>>>(x, Wt, b_img, h);
  k2_scores <<<N_INST / 128, 256, 0, stream>>>(h, cbf, cnorm, w_attn, b_attn, g1,
                                               pval, pidx);
  k3_argmax <<<1, 64, 0, stream>>>(pval, pidx, N_INST / 128, idx);
  k4_rep_f1 <<<CNUM, 256, 0, stream>>>(h, idx, W1t, rep, repn, f1);
  k5_graph  <<<1, 256, 0, stream>>>(rep, repn, g2, f1, al1, ar1, b1, W2, al2, ar2,
                                    b2, Wc, bc, (float*)d_out);
}
